// HPDecoder_84705345011916
// MI455X (gfx1250) — compile-verified, matrix operand order checked by reading, not run on hardware
//
#include <hip/hip_runtime.h>
#include <hip/hip_bf16.h>

typedef __attribute__((ext_vector_type(16))) _Float16 v16h;
typedef __attribute__((ext_vector_type(8)))  _Float16 v8h;
typedef __attribute__((ext_vector_type(8)))  float    v8f;

// ---------------------------------------------------------------------------
// helpers
// ---------------------------------------------------------------------------
__device__ __forceinline__ unsigned f2key(float f) {
  // order-preserving float -> uint mapping (larger float -> larger key)
  unsigned u = __float_as_uint(f);
  return (u & 0x80000000u) ? ~u : (u | 0x80000000u);
}

// ---------------------------------------------------------------------------
// stage-0 input: f32 (32^3,32) -> f16 halo-padded (34^3,32)
// ---------------------------------------------------------------------------
__global__ void cvt_x_padded(const float* __restrict__ in,
                             _Float16* __restrict__ out) {
  int v = blockIdx.x * blockDim.x + threadIdx.x;
  if (v >= 32 * 32 * 32) return;
  const int x = v & 31, y = (v >> 5) & 31, z = v >> 10;
  const float* src = in + (size_t)v * 32;
  _Float16* dst = out + ((size_t)((z + 1) * 34 + (y + 1)) * 34 + (x + 1)) * 32;
#pragma unroll
  for (int c = 0; c < 32; ++c) dst[c] = (_Float16)src[c];
}

// ---------------------------------------------------------------------------
// Pre-pack conv weights (f32, layout (27,CIN,16)) into the WMMA B-fragment
// lane layout in f16:  bpack[chunk][lane][16 halves].
//   B (16-bit 32x16): lane L -> col N=L&15; lanes 0-15 carry K 0..7 & 16..23,
//   lanes 16-31 carry K 8..15 & 24..31.
//   CIN=32: chunk c == tap c, K = input channel.
//   CIN=16: chunk c packs taps 2c (K 0..15) and 2c+1 (K 16..31), tap 27 -> 0.
// ---------------------------------------------------------------------------
template <int CIN>
__global__ void pack_weights(const float* __restrict__ w,
                             _Float16* __restrict__ bpack) {
  constexpr int NCHUNK = (CIN == 32) ? 27 : 14;
  int t = blockIdx.x * blockDim.x + threadIdx.x;
  if (t >= NCHUNK * 32) return;
  const int c = t >> 5, lane = t & 31;
  const int hi = lane >> 4, n = lane & 15;
  _Float16 vals[16];
  if (CIN == 32) {
    const int ta = c;
    const int kb0 = hi ? 8 : 0;
#pragma unroll
    for (int j = 0; j < 8; ++j) {
      vals[j]     = (_Float16)w[((ta * 32 + kb0 + j) * 16) + n];
      vals[j + 8] = (_Float16)w[((ta * 32 + kb0 + 16 + j) * 16) + n];
    }
  } else {
    const int ta = 2 * c, tb = 2 * c + 1;
    const int kb = hi ? 8 : 0;
#pragma unroll
    for (int j = 0; j < 8; ++j) {
      vals[j]     = (_Float16)w[((ta * 16 + kb + j) * 16) + n];
      vals[j + 8] = (tb < 27) ? (_Float16)w[((tb * 16 + kb + j) * 16) + n]
                              : (_Float16)0.f;
    }
  }
#pragma unroll
  for (int j = 0; j < 16; ++j) bpack[(size_t)t * 16 + j] = vals[j];
}

// ---------------------------------------------------------------------------
// 3x3x3 conv, Cin in {16,32}, Cout=16, SAME padding via halo-padded input,
// occ-mask + bias + relu.  One wave32 computes a 16-voxel x 16-channel tile
// with v_wmma_f32_16x16x32_f16.
//   Cin=32: one tap == one K=32 WMMA (27 WMMAs / tile)
//   Cin=16: two taps packed per K=32 WMMA (14 WMMAs / tile)
// All A loads are unconditional b128s off ONE base address with compile-time
// immediate offsets (halo guarantees in-bounds); B fragments are pre-packed.
// Epilogue transposes through a per-wave LDS tile -> two b128 stores/lane.
// ---------------------------------------------------------------------------
template <int LOGD, int CIN>
__global__ void conv3x3x3_wmma(const _Float16* __restrict__ inp,  // (D+2)^3*CIN
                               const _Float16* __restrict__ bpack,
                               const float* __restrict__ bias,    // (16)
                               const unsigned char* __restrict__ occ,
                               float* __restrict__ out)           // (D^3,16)
{
  constexpr int D = 1 << LOGD;
  constexpr int DP = D + 2;
  __shared__ float tile[8][256];     // 8 waves per 256-thread block

  const int lane = threadIdx.x & 31;
  const int wib  = (threadIdx.x >> 5) & 7;
  const int wave = (blockIdx.x * blockDim.x + threadIdx.x) >> 5;
  const int base = wave * 16;        // first voxel of this tile
  const int hi   = lane >> 4;        // 0: low K half, 1: high K half
  const int n    = lane & 15;        // A row M and B column N

  const int mvox = base + n;         // this lane's A-row voxel
  const int x0 = mvox & (D - 1);
  const int y0 = (mvox >> LOGD) & (D - 1);
  const int z0 = mvox >> (2 * LOGD);

  // one base pointer; every tap is a compile-time immediate offset from it
  const _Float16* pbase =
      inp + ((size_t)((z0 + 1) * DP + (y0 + 1)) * DP + (x0 + 1)) * CIN +
      (hi ? 8 : 0);
  const _Float16* bbase = bpack + (size_t)lane * 16;

  v8f acc = {};
  constexpr int NCHUNK = (CIN == 32) ? 27 : 14;

#pragma unroll
  for (int c = 0; c < NCHUNK; ++c) {
    const int ta = (CIN == 32) ? c : 2 * c;
    const int tb = (CIN == 32) ? c : 2 * c + 1;
    const int offa =
        (((ta / 9 - 1) * DP + ((ta % 9) / 3 - 1)) * DP + (ta % 3 - 1)) * CIN;

    v8h alo = *(const v8h*)(pbase + offa);
    v8h ahi;
    if (CIN == 32) {
      ahi = *(const v8h*)(pbase + offa + 16);
    } else if (tb < 27) {
      const int offb =
          (((tb / 9 - 1) * DP + ((tb % 9) / 3 - 1)) * DP + (tb % 3 - 1)) * CIN;
      ahi = *(const v8h*)(pbase + offb);
    } else {
      ahi = (v8h){};
    }
    v16h a = __builtin_shufflevector(alo, ahi, 0, 1, 2, 3, 4, 5, 6, 7, 8, 9, 10,
                                     11, 12, 13, 14, 15);

    v8h blo = *(const v8h*)(bbase + (size_t)c * 512);
    v8h bhi = *(const v8h*)(bbase + (size_t)c * 512 + 8);
    v16h bf = __builtin_shufflevector(blo, bhi, 0, 1, 2, 3, 4, 5, 6, 7, 8, 9,
                                      10, 11, 12, 13, 14, 15);

    acc = __builtin_amdgcn_wmma_f32_16x16x32_f16(false, a, false, bf, (short)0,
                                                 acc, false, false);
  }

  // C/D layout: VGPR r -> M=r (lanes 0-15) / M=r+8 (lanes 16-31), N = lane&15
  const float bn = bias[n];
  float* t = tile[wib];
#pragma unroll
  for (int r = 0; r < 8; ++r) {
    const int vr = r + hi * 8;       // voxel within tile, 0..15
    float v = acc[r] + bn;
    v = occ[base + vr] ? fmaxf(v, 0.f) : 0.f;
    t[vr * 16 + n] = v;
  }
  __builtin_amdgcn_wave_barrier();   // wave-local LDS transpose, no WG barrier

  const int vL = lane >> 1;          // voxel handled by this lane
  const int hL = (lane & 1) * 8;     // which 8-channel half
  const float* src = t + vL * 16 + hL;
  float4 p0 = *(const float4*)(src);
  float4 p1 = *(const float4*)(src + 4);
  float* dst = out + (size_t)(base + vL) * 16 + hL;
  *(float4*)(dst)     = p0;
  *(float4*)(dst + 4) = p1;
}

// ---------------------------------------------------------------------------
// classifier conv: 3x3x3, 16 -> 1, bias, masked by occ (scalar per voxel)
// ---------------------------------------------------------------------------
template <int LOGD>
__global__ void cls_conv3(const float* __restrict__ feat,
                          const float* __restrict__ w,  // (27,16)
                          const float* __restrict__ b,  // (1)
                          const unsigned char* __restrict__ occ,
                          float* __restrict__ out) {
  constexpr int D = 1 << LOGD;
  int i = blockIdx.x * blockDim.x + threadIdx.x;
  if (i >= D * D * D) return;
  if (!occ[i]) { out[i] = 0.f; return; }
  const int x = i & (D - 1), y = (i >> LOGD) & (D - 1), z = i >> (2 * LOGD);
  float acc = b[0];
  for (int t = 0; t < 27; ++t) {
    const int dz = t / 9 - 1, dy = (t % 9) / 3 - 1, dx = t % 3 - 1;
    const int zz = z + dz, yy = y + dy, xx = x + dx;
    if ((unsigned)zz >= (unsigned)D || (unsigned)yy >= (unsigned)D ||
        (unsigned)xx >= (unsigned)D)
      continue;
    const float* f = feat + (size_t)((zz * D + yy) * D + xx) * 16;
    const float* wp = w + t * 16;
#pragma unroll
    for (int c = 0; c < 16; ++c) acc += f[c] * wp[c];
  }
  out[i] = acc;
}

// ---------------------------------------------------------------------------
// k-th largest (radix select over order-preserving keys) among occupied voxels
// single 1024-thread block, 4 passes of 8 bits, deterministic
// ---------------------------------------------------------------------------
__global__ void kth_select(const float* __restrict__ logits,
                           const unsigned char* __restrict__ occ, int n,
                           const int* __restrict__ kptr,
                           unsigned int* __restrict__ thr_out) {
  __shared__ unsigned int hist[256];
  __shared__ unsigned int prefix_s, krem_s;
  const int tid = threadIdx.x;
  if (tid == 0) {
    int k = *kptr;
    prefix_s = 0u;
    krem_s = (k < 1) ? 1u : (unsigned)k;
  }
  __syncthreads();
  for (int pass = 0; pass < 4; ++pass) {
    if (tid < 256) hist[tid] = 0u;
    __syncthreads();
    const unsigned prefix = prefix_s;
    const int shift = 24 - 8 * pass;
    const unsigned mask_hi = (pass == 0) ? 0u : (0xFFFFFFFFu << (shift + 8));
    for (int i = tid; i < n; i += 1024) {
      if (!occ[i]) continue;
      const unsigned key = f2key(logits[i]);
      if ((key & mask_hi) == (prefix & mask_hi))
        atomicAdd(&hist[(key >> shift) & 0xFFu], 1u);
    }
    __syncthreads();
    if (tid == 0) {
      unsigned krem = krem_s, cum = 0;
      int bsel = 0;
      for (int b = 255; b >= 0; --b) {
        if (cum + hist[b] >= krem) { bsel = b; break; }
        cum += hist[b];
        if (b == 0) bsel = 0;  // fewer than k occupied: take lowest bucket
      }
      krem_s = (krem > cum) ? (krem - cum) : 1u;
      prefix_s = prefix | ((unsigned)bsel << shift);
    }
    __syncthreads();
  }
  if (tid == 0) *thr_out = prefix_s;
}

// ---------------------------------------------------------------------------
// prune: mask = (logit>=thr & occ) | (gt & occ); zero pruned features
// ---------------------------------------------------------------------------
__global__ void apply_mask(float* __restrict__ feats,
                           const float* __restrict__ logits,
                           const unsigned char* __restrict__ occ,
                           const unsigned char* __restrict__ gt,
                           const unsigned int* __restrict__ thr,
                           unsigned char* __restrict__ mask_out, int n) {
  int i = blockIdx.x * blockDim.x + threadIdx.x;
  if (i >= n) return;
  const unsigned t = *thr;
  const bool keep = occ[i] && (gt[i] || (f2key(logits[i]) >= t));
  mask_out[i] = keep ? 1 : 0;
  if (!keep) {
#pragma unroll
    for (int c = 0; c < 16; ++c) feats[(size_t)i * 16 + c] = 0.f;
  }
}

// ---------------------------------------------------------------------------
// 2x transpose-conv upsample (2,2,2 kernel, stride 2, VALID): each output
// voxel has exactly one parent and one weight slot.  Fuses +bias, occ-up mask
// and the following relu; writes f16 activations into the halo-padded buffer
// for the next WMMA conv (halo + pruned interiors are zero via memset).
// ---------------------------------------------------------------------------
template <int LOGDIN>
__global__ void up2x_relu(const float* __restrict__ feat,          // (DIN^3,16)
                          const unsigned char* __restrict__ maskp, // (DIN^3)
                          const float* __restrict__ w,             // (8,16,16)
                          const float* __restrict__ b,             // (16)
                          _Float16* __restrict__ outh,             // (D+2)^3*16
                          unsigned char* __restrict__ occ_out) {
  constexpr int DIN = 1 << LOGDIN;
  constexpr int LOGOUT = LOGDIN + 1;
  constexpr int D = DIN * 2;
  constexpr int DP = D + 2;
  int i = blockIdx.x * blockDim.x + threadIdx.x;
  if (i >= D * D * D) return;
  const int x = i & (D - 1), y = (i >> LOGOUT) & (D - 1), z = i >> (2 * LOGOUT);
  const int parent = ((z >> 1) * DIN + (y >> 1)) * DIN + (x >> 1);
  const unsigned char m = maskp[parent];
  occ_out[i] = m;
  if (!m) return;  // padded buffer pre-zeroed each call
  const int slot = ((z & 1) * 2 + (y & 1)) * 2 + (x & 1);
  const float* wp = w + slot * 256;
  float acc[16];
#pragma unroll
  for (int c = 0; c < 16; ++c) acc[c] = b[c];
  for (int ci = 0; ci < 16; ++ci) {
    const float v = feat[(size_t)parent * 16 + ci];
#pragma unroll
    for (int c = 0; c < 16; ++c) acc[c] += v * wp[ci * 16 + c];
  }
  _Float16* dst =
      outh + ((size_t)((z + 1) * DP + (y + 1)) * DP + (x + 1)) * 16;
#pragma unroll
  for (int c = 0; c < 16; ++c) dst[c] = (_Float16)fmaxf(acc[c], 0.f);
}

// ---------------------------------------------------------------------------
// launch
// ---------------------------------------------------------------------------
extern "C" void kernel_launch(void* const* d_in, const int* in_sizes, int n_in,
                              void* d_out, int out_size, void* d_ws,
                              size_t ws_size, hipStream_t stream) {
  (void)in_sizes; (void)n_in; (void)out_size; (void)ws_size;

  const float*         x       = (const float*)d_in[0];
  const unsigned char* occ0    = (const unsigned char*)d_in[1];
  const unsigned char* gt0     = (const unsigned char*)d_in[2];
  const unsigned char* gt1     = (const unsigned char*)d_in[3];
  const unsigned char* gt2     = (const unsigned char*)d_in[4];
  const float* w_conv0 = (const float*)d_in[5];
  const float* b_conv0 = (const float*)d_in[6];
  const float* w_cls0  = (const float*)d_in[7];
  const float* b_cls0  = (const float*)d_in[8];
  const float* w_up1   = (const float*)d_in[9];
  const float* b_up1   = (const float*)d_in[10];
  const float* w_conv1 = (const float*)d_in[11];
  const float* b_conv1 = (const float*)d_in[12];
  const float* w_cls1  = (const float*)d_in[13];
  const float* b_cls1  = (const float*)d_in[14];
  const float* w_up2   = (const float*)d_in[15];
  const float* b_up2   = (const float*)d_in[16];
  const float* w_conv2 = (const float*)d_in[17];
  const float* b_conv2 = (const float*)d_in[18];
  const float* w_cls2  = (const float*)d_in[19];
  const float* b_cls2  = (const float*)d_in[20];
  const int* nums0 = (const int*)d_in[21];
  const int* nums1 = (const int*)d_in[22];
  const int* nums2 = (const int*)d_in[23];

  constexpr int N0 = 32 * 32 * 32;       // 32768
  constexpr int N1 = 64 * 64 * 64;       // 262144
  constexpr int N2 = 128 * 128 * 128;    // 2097152
  constexpr size_t P0 = (size_t)34 * 34 * 34 * 32;    // padded stage-0 halves
  constexpr size_t P1 = (size_t)66 * 66 * 66 * 16;    // padded stage-1 halves
  constexpr size_t P2 = (size_t)130 * 130 * 130 * 16; // padded stage-2 halves

  // workspace layout (all offsets 256B-aligned)
  char* ws = (char*)d_ws;
  size_t off = 0;
  auto alloc = [&](size_t bytes) {
    size_t o = off;
    off += (bytes + 255) & ~(size_t)255;
    return o;
  };
  _Float16* xhp   = (_Float16*)(ws + alloc(P0 * 2));
  float*    feat0 = (float*)   (ws + alloc((size_t)N0 * 16 * 4));
  unsigned char* mask0 = (unsigned char*)(ws + alloc(N0));
  _Float16* h1p   = (_Float16*)(ws + alloc(P1 * 2));
  unsigned char* occ1  = (unsigned char*)(ws + alloc(N1));
  float*    feat1 = (float*)   (ws + alloc((size_t)N1 * 16 * 4));
  unsigned char* mask1 = (unsigned char*)(ws + alloc(N1));
  _Float16* h2p   = (_Float16*)(ws + alloc(P2 * 2));
  unsigned char* occ2  = (unsigned char*)(ws + alloc(N2));
  unsigned char* mask2 = (unsigned char*)(ws + alloc(N2));
  _Float16* bpk0 = (_Float16*)(ws + alloc((size_t)27 * 32 * 16 * 2));
  _Float16* bpk1 = (_Float16*)(ws + alloc((size_t)14 * 32 * 16 * 2));
  _Float16* bpk2 = (_Float16*)(ws + alloc((size_t)14 * 32 * 16 * 2));
  unsigned int* thr = (unsigned int*)(ws + alloc(3 * sizeof(unsigned int)));

  // output layout: cls0 | cls1 | cls2 | out(128^3 x 16)
  float* cls0 = (float*)d_out;
  float* cls1 = cls0 + N0;
  float* cls2 = cls1 + N1;
  float* outF = cls2 + N2;

  // ---- zero the halo-padded activation buffers (capture-safe async memset)
  hipMemsetAsync(xhp, 0, P0 * 2, stream);
  hipMemsetAsync(h1p, 0, P1 * 2, stream);
  hipMemsetAsync(h2p, 0, P2 * 2, stream);

  // ---- weight pre-pack (tiny, once per call) ----
  pack_weights<32><<<(27 * 32 + 255) / 256, 256, 0, stream>>>(w_conv0, bpk0);
  pack_weights<16><<<(14 * 32 + 255) / 256, 256, 0, stream>>>(w_conv1, bpk1);
  pack_weights<16><<<(14 * 32 + 255) / 256, 256, 0, stream>>>(w_conv2, bpk2);

  // ---- stage 0 (32^3, Cin=32) ----
  cvt_x_padded<<<(N0 + 255) / 256, 256, 0, stream>>>(x, xhp);
  conv3x3x3_wmma<5, 32><<<N0 / 128, 256, 0, stream>>>(xhp, bpk0, b_conv0,
                                                      occ0, feat0);
  cls_conv3<5><<<N0 / 256, 256, 0, stream>>>(feat0, w_cls0, b_cls0, occ0, cls0);
  kth_select<<<1, 1024, 0, stream>>>(cls0, occ0, N0, nums0, thr + 0);
  apply_mask<<<N0 / 256, 256, 0, stream>>>(feat0, cls0, occ0, gt0, thr + 0,
                                           mask0, N0);

  // ---- stage 1 (64^3, Cin=16) ----
  up2x_relu<5><<<N1 / 256, 256, 0, stream>>>(feat0, mask0, w_up1, b_up1, h1p,
                                             occ1);
  conv3x3x3_wmma<6, 16><<<N1 / 128, 256, 0, stream>>>(h1p, bpk1, b_conv1,
                                                      occ1, feat1);
  cls_conv3<6><<<N1 / 256, 256, 0, stream>>>(feat1, w_cls1, b_cls1, occ1, cls1);
  kth_select<<<1, 1024, 0, stream>>>(cls1, occ1, N1, nums1, thr + 1);
  apply_mask<<<N1 / 256, 256, 0, stream>>>(feat1, cls1, occ1, gt1, thr + 1,
                                           mask1, N1);

  // ---- stage 2 (128^3, Cin=16); features written straight into d_out ----
  up2x_relu<6><<<N2 / 256, 256, 0, stream>>>(feat1, mask1, w_up2, b_up2, h2p,
                                             occ2);
  conv3x3x3_wmma<7, 16><<<N2 / 128, 256, 0, stream>>>(h2p, bpk2, b_conv2,
                                                      occ2, outF);
  cls_conv3<7><<<N2 / 256, 256, 0, stream>>>(outF, w_cls2, b_cls2, occ2, cls2);
  kth_select<<<1, 1024, 0, stream>>>(cls2, occ2, N2, nums2, thr + 2);
  apply_mask<<<N2 / 256, 256, 0, stream>>>(outF, cls2, occ2, gt2, thr + 2,
                                           mask2, N2);
}
